// MS_Attention_linear_3650722202336
// MI455X (gfx1250) — 16-bit and 8-bit matrix operand order repaired and checked by reading, compiled, not run on hardware
//
#include <hip/hip_runtime.h>
#include <hip/hip_bf16.h>
#include <math.h>

typedef __attribute__((ext_vector_type(16))) _Float16 v16h;
typedef __attribute__((ext_vector_type(8)))  _Float16 v8h;
typedef __attribute__((ext_vector_type(8)))  float    v8f;
typedef __attribute__((ext_vector_type(8)))  int      v8i;
typedef __attribute__((ext_vector_type(4)))  _Float16 v4h;

#define C_DIM 384
#define N_PIX 4096
#define NB    8
#define NH    8
#define HD    48
#define BN_EPS 1e-5f
#define WPITCH 392

__device__ __forceinline__ float quant_clip(float y) {
    return rintf(fminf(fmaxf(y, 0.f), 4.f));
}

__device__ __forceinline__ v16h load_a_f16_lds(const _Float16* __restrict__ Wl,
                                               int k0, int lane) {
    int m  = lane & 15;
    int kb = k0 + ((lane >> 4) << 4);
    union { v16h v; uint4 q[2]; } r;
    const _Float16* row = Wl + m * WPITCH + kb;
    r.q[0] = *(const uint4*)(row);
    r.q[1] = *(const uint4*)(row + 8);
    return r.v;
}

__device__ __forceinline__ v16h load_a_f16_ld64(const _Float16* __restrict__ A,
                                                int m0, int k0, int lane) {
    int m  = m0 + (lane & 15);
    int kb = (lane >> 4) << 4;
    union { v16h v; unsigned u[8]; } r;
    const char* p = (const char*)(A + (size_t)m * 64 + k0 + kb);
#pragma unroll
    for (int i = 0; i < 8; ++i) {
        int off = i << 1;
        r.u[i] = *(const unsigned*)(p + 2 * off);
    }
    return r.v;
}

__device__ __forceinline__ v16h load_b_f16_chlast(const _Float16* __restrict__ X,
                                                  int ncol, int k0, int lane) {
    int n  = ncol + (lane & 15);
    int kb = k0 + ((lane >> 4) << 4);
    union { v16h v; unsigned u[8]; } r;
    const unsigned* p = (const unsigned*)(X + (size_t)n * C_DIM + kb);
#pragma unroll
    for (int j = 0; j < 8; ++j) r.u[j] = p[j];
    return r.v;
}

__device__ __forceinline__ v16h load_b_f16_qt(const _Float16* __restrict__ Qt,
                                              int hoff, int ncol, int k0, int lane) {
    int n  = ncol + (lane & 15);
    int kb = k0 + ((lane >> 4) << 4);
    union { v16h v; unsigned u[8]; } r;
    const unsigned* p = (const unsigned*)(Qt + (size_t)n * C_DIM + hoff + kb);
#pragma unroll
    for (int j = 0; j < 8; ++j) r.u[j] = (kb + 2 * j < HD) ? p[j] : 0u;
    return r.v;
}

__device__ __forceinline__ v8i load_a_iu8(const unsigned char* __restrict__ A,
                                          int m0, int k0, int lane) {
    int m  = m0 + (lane & 15);
    int kb = (lane >> 4) << 4;
    union { v8i v; unsigned u[8]; } r;
    const unsigned char* p = A + (size_t)m * N_PIX + k0 + kb;
#pragma unroll
    for (int j = 0; j < 8; ++j) {
        int off = ((j >> 2) << 5) + ((j & 3) << 2);
        r.u[j] = *(const unsigned*)(p + off);
    }
    return r.v;
}

__device__ __forceinline__ v8i load_b_iu8(const unsigned char* __restrict__ Bm,
                                          int n0, int k0, int lane) {
    int n  = n0 + (lane & 15);
    int kb = (lane >> 4) << 4;
    union { v8i v; unsigned u[8]; } r;
    const unsigned char* p = Bm + (size_t)n * N_PIX + k0 + kb;
#pragma unroll
    for (int j = 0; j < 8; ++j) {
        int off = ((j >> 2) << 5) + ((j & 3) << 2);
        r.u[j] = *(const unsigned*)(p + off);
    }
    return r.v;
}

__device__ __forceinline__ void stage_wstrip_async(const _Float16* __restrict__ W, int m0,
                                                   _Float16* ldsdst, int tid) {
#pragma unroll
    for (int t = 0; t < 3; ++t) {
        int e = t * 256 + tid;
        int m = e / 48, s = e % 48;
        unsigned loff = (unsigned)(size_t)(ldsdst + m * WPITCH + s * 8);
        unsigned long long g =
            (unsigned long long)(size_t)(W + (size_t)(m0 + m) * C_DIM + s * 8);
        asm volatile("global_load_async_to_lds_b128 %0, %1, off"
                     :: "v"(loff), "v"(g) : "memory");
    }
}

__global__ void bn_scales_kernel(const float* __restrict__ g, const float* __restrict__ be,
                                 const float* __restrict__ mn, const float* __restrict__ va,
                                 float* __restrict__ out) {
    int c = blockIdx.x * blockDim.x + threadIdx.x;
    if (c < C_DIM) {
        float s = g[c] * rsqrtf(va[c] + BN_EPS);
        out[c]         = s;
        out[C_DIM + c] = be[c] - mn[c] * s;
    }
}

__global__ __launch_bounds__(256) void wcvt_kernel(const float* __restrict__ w,
                                                   _Float16* __restrict__ wh) {
    int i = blockIdx.x * 256 + threadIdx.x;
    float4 v = ((const float4*)w)[i];
    v4h o;
    o[0] = (_Float16)v.x; o[1] = (_Float16)v.y;
    o[2] = (_Float16)v.z; o[3] = (_Float16)v.w;
    ((v4h*)wh)[i] = o;
}

__global__ __launch_bounds__(256) void quant_tr_kernel(const float* __restrict__ x,
                                                       _Float16* __restrict__ xs_t) {
    __shared__ _Float16 tile[32][65];
    int b = blockIdx.z, c0 = blockIdx.y * 32, n0 = blockIdx.x * 64;
    int tid = threadIdx.x;
    const float* xb = x + ((size_t)b * C_DIM + c0) * N_PIX;
#pragma unroll
    for (int t = 0; t < 8; ++t) {
        int e = t * 256 + tid;
        int ci = e >> 6, nj = e & 63;
        float v = xb[(size_t)ci * N_PIX + n0 + nj];
        tile[ci][nj] = (_Float16)(quant_clip(v) * 0.25f);
    }
    __syncthreads();
    _Float16* ob = xs_t + ((size_t)b * N_PIX + n0) * C_DIM + c0;
#pragma unroll
    for (int t = 0; t < 8; ++t) {
        int e = t * 256 + tid;
        int ci = e & 31, nj = e >> 5;
        ob[(size_t)nj * C_DIM + ci] = tile[ci][nj];
    }
}

__global__ __launch_bounds__(256) void qkv_gemm_kernel(
    const _Float16* __restrict__ xs_t,
    const _Float16* __restrict__ qw, const _Float16* __restrict__ kw,
    const _Float16* __restrict__ vw, const float* __restrict__ scales,
    _Float16* __restrict__ qt, unsigned char* __restrict__ k8, unsigned char* __restrict__ v8o) {
    __shared__ _Float16 wl[3 * 16 * WPITCH];
    int b = blockIdx.z, m0 = blockIdx.y * 16;
    int tid = threadIdx.x, lane = tid & 31, wave = tid >> 5;
    int n0 = blockIdx.x * 256 + wave * 32;
    const _Float16* xsb = xs_t + (size_t)b * N_PIX * C_DIM;

    stage_wstrip_async(qw, m0, wl, tid);
    stage_wstrip_async(kw, m0, wl + 16 * WPITCH, tid);
    stage_wstrip_async(vw, m0, wl + 32 * WPITCH, tid);
    asm volatile("s_wait_asynccnt 0x0" ::: "memory");
    __syncthreads();

    v8f cq0 = {}, cq1 = {}, ck0 = {}, ck1 = {}, cv0 = {}, cv1 = {};
    for (int k0 = 0; k0 < C_DIM; k0 += 32) {
        v16h b0 = load_b_f16_chlast(xsb, n0, k0, lane);
        v16h b1 = load_b_f16_chlast(xsb, n0 + 16, k0, lane);
        v16h aq = load_a_f16_lds(wl, k0, lane);
        v16h ak = load_a_f16_lds(wl + 16 * WPITCH, k0, lane);
        v16h av = load_a_f16_lds(wl + 32 * WPITCH, k0, lane);
        cq0 = __builtin_amdgcn_wmma_f32_16x16x32_f16(false, aq, false, b0, (short)0, cq0, false, false);
        cq1 = __builtin_amdgcn_wmma_f32_16x16x32_f16(false, aq, false, b1, (short)0, cq1, false, false);
        ck0 = __builtin_amdgcn_wmma_f32_16x16x32_f16(false, ak, false, b0, (short)0, ck0, false, false);
        ck1 = __builtin_amdgcn_wmma_f32_16x16x32_f16(false, ak, false, b1, (short)0, ck1, false, false);
        cv0 = __builtin_amdgcn_wmma_f32_16x16x32_f16(false, av, false, b0, (short)0, cv0, false, false);
        cv1 = __builtin_amdgcn_wmma_f32_16x16x32_f16(false, av, false, b1, (short)0, cv1, false, false);
    }

    int rb = (lane >> 4) << 3;
    const float* sq = scales;
    const float* sk = scales + 2 * C_DIM;
    const float* sv = scales + 4 * C_DIM;
    v8f aq[2] = {cq0, cq1}, ak[2] = {ck0, ck1}, av[2] = {cv0, cv1};
#pragma unroll
    for (int t = 0; t < 2; ++t) {
        int n = n0 + t * 16 + (lane & 15);
        union { v8h v; _Float16 h[8]; } oq;
#pragma unroll
        for (int r = 0; r < 8; ++r) {
            int c = m0 + rb + r;
            float y = aq[t][r] * sq[c] + sq[C_DIM + c];
            oq.h[r] = (_Float16)(quant_clip(y) * 0.25f);
            y = ak[t][r] * sk[c] + sk[C_DIM + c];
            k8[((size_t)b * C_DIM + c) * N_PIX + n] = (unsigned char)quant_clip(y);
            y = av[t][r] * sv[c] + sv[C_DIM + c];
            v8o[((size_t)b * C_DIM + c) * N_PIX + n] = (unsigned char)quant_clip(y);
        }
        *(v8h*)(qt + ((size_t)b * N_PIX + n) * C_DIM + m0 + rb) = oq.v;
    }
}

__global__ __launch_bounds__(32) void kv_gemm_kernel(const unsigned char* __restrict__ k8,
                                                     const unsigned char* __restrict__ v8,
                                                     _Float16* __restrict__ kvT) {
    int tile = blockIdx.x, bh = blockIdx.y;
    int d0 = (tile / 3) * 16, e0 = (tile % 3) * 16;
    int lane = threadIdx.x;
    const unsigned char* kb = k8 + (size_t)bh * HD * N_PIX;
    const unsigned char* vb = v8 + (size_t)bh * HD * N_PIX;

    v8i acc = {};
    for (int k0 = 0; k0 < N_PIX; k0 += 64) {
        v8i a = load_a_iu8(kb, d0, k0, lane);
        v8i b = load_b_iu8(vb, e0, k0, lane);
        acc = __builtin_amdgcn_wmma_i32_16x16x64_iu8(false, a, false, b, acc, false, false);
    }
    const float KV_SCALE = 2.0f / (16.0f * sqrtf((float)HD));
    int e  = e0 + (lane & 15);
    int rb = (lane >> 4) << 3;
    _Float16* row = kvT + ((size_t)bh * HD + e) * 64;
#pragma unroll
    for (int r = 0; r < 8; ++r)
        row[d0 + rb + r] = (_Float16)((float)acc[r] * KV_SCALE);
    if (d0 == 32) {
#pragma unroll
        for (int r = 0; r < 8; ++r) row[48 + rb + r] = (_Float16)0.f;
    }
}

__global__ __launch_bounds__(256) void attn_gemm_kernel(const _Float16* __restrict__ qt,
                                                        const _Float16* __restrict__ kvT,
                                                        _Float16* __restrict__ a_t) {
    int bh = blockIdx.y;
    int b = bh >> 3, h = bh & 7;
    int lane = threadIdx.x & 31, wave = threadIdx.x >> 5;
    int gw = blockIdx.x * 8 + wave;
    int mt = gw / 64, nb = gw % 64;
    int e0 = mt * 16, n0 = nb * 64;

    const _Float16* kvb = kvT + (size_t)bh * HD * 64;
    const _Float16* qb  = qt + (size_t)b * N_PIX * C_DIM;

    v8f acc[4] = {{}, {}, {}, {}};
#pragma unroll
    for (int k0 = 0; k0 < 64; k0 += 32) {
        v16h a = load_a_f16_ld64(kvb, e0, k0, lane);
#pragma unroll
        for (int t = 0; t < 4; ++t) {
            v16h bf = load_b_f16_qt(qb, h * HD, n0 + t * 16, k0, lane);
            acc[t] = __builtin_amdgcn_wmma_f32_16x16x32_f16(false, a, false, bf, (short)0,
                                                            acc[t], false, false);
        }
    }
    int rb = (lane >> 4) << 3;
#pragma unroll
    for (int t = 0; t < 4; ++t) {
        int n = n0 + t * 16 + (lane & 15);
        union { v8h v; _Float16 h[8]; } o;
#pragma unroll
        for (int r = 0; r < 8; ++r)
            o.h[r] = (_Float16)(quant_clip(acc[t][r]) * 0.25f);
        *(v8h*)(a_t + ((size_t)b * N_PIX + n) * C_DIM + h * HD + e0 + rb) = o.v;
    }
}

__global__ __launch_bounds__(256) void pconv_gemm_kernel(const _Float16* __restrict__ a_t,
                                                         const _Float16* __restrict__ pw,
                                                         const float* __restrict__ sc,
                                                         float* __restrict__ out) {
    __shared__ _Float16 wl[16 * WPITCH];
    int b = blockIdx.z, m0 = blockIdx.y * 16;
    int tid = threadIdx.x, lane = tid & 31, wave = tid >> 5;
    int n0 = blockIdx.x * 256 + wave * 32;
    const _Float16* ab = a_t + (size_t)b * N_PIX * C_DIM;

    stage_wstrip_async(pw, m0, wl, tid);
    asm volatile("s_wait_asynccnt 0x0" ::: "memory");
    __syncthreads();

    v8f c0 = {}, c1 = {};
    for (int k0 = 0; k0 < C_DIM; k0 += 32) {
        v16h b0 = load_b_f16_chlast(ab, n0, k0, lane);
        v16h b1 = load_b_f16_chlast(ab, n0 + 16, k0, lane);
        v16h af = load_a_f16_lds(wl, k0, lane);
        c0 = __builtin_amdgcn_wmma_f32_16x16x32_f16(false, af, false, b0, (short)0, c0, false, false);
        c1 = __builtin_amdgcn_wmma_f32_16x16x32_f16(false, af, false, b1, (short)0, c1, false, false);
    }
    int rb = (lane >> 4) << 3;
    v8f cc[2] = {c0, c1};
#pragma unroll
    for (int t = 0; t < 2; ++t) {
        int n = n0 + t * 16 + (lane & 15);
#pragma unroll
        for (int r = 0; r < 8; ++r) {
            int c = m0 + rb + r;
            out[((size_t)b * C_DIM + c) * N_PIX + n] = cc[t][r] * sc[c] + sc[C_DIM + c];
        }
    }
}

extern "C" void kernel_launch(void* const* d_in, const int* in_sizes, int n_in,
                              void* d_out, int out_size, void* d_ws, size_t ws_size,
                              hipStream_t stream) {
    (void)in_sizes; (void)n_in; (void)out_size; (void)ws_size;
    const float* x  = (const float*)d_in[0];
    const float* qw = (const float*)d_in[1];
    const float* qg = (const float*)d_in[2], *qb = (const float*)d_in[3];
    const float* qm = (const float*)d_in[4], *qv = (const float*)d_in[5];
    const float* kw = (const float*)d_in[6];
    const float* kg = (const float*)d_in[7], *kb = (const float*)d_in[8];
    const float* km = (const float*)d_in[9], *kv = (const float*)d_in[10];
    const float* vw = (const float*)d_in[11];
    const float* vg = (const float*)d_in[12], *vb = (const float*)d_in[13];
    const float* vm = (const float*)d_in[14], *vv = (const float*)d_in[15];
    const float* pw = (const float*)d_in[16];
    const float* pg = (const float*)d_in[17], *pb = (const float*)d_in[18];
    const float* pm = (const float*)d_in[19], *pv = (const float*)d_in[20];

    char* ws = (char*)d_ws;
    const size_t SZ_H  = (size_t)NB * C_DIM * N_PIX * sizeof(_Float16);
    const size_t SZ_B  = (size_t)NB * C_DIM * N_PIX;
    const size_t SZ_KV = (size_t)NB * NH * HD * 64 * sizeof(_Float16);
    const size_t SZ_SCL = 4 * 2 * C_DIM * sizeof(float);

    _Float16*      xs_t = (_Float16*)(ws);
    _Float16*      qt   = (_Float16*)(ws + SZ_H);
    unsigned char* k8   = (unsigned char*)(ws + 2 * SZ_H);
    unsigned char* v8p  = (unsigned char*)(ws + 2 * SZ_H + SZ_B);
    _Float16*      kvT  = (_Float16*)(ws + 2 * SZ_H + 2 * SZ_B);
    float*         scl  = (float*)(ws + 2 * SZ_H + 2 * SZ_B + SZ_KV);
    _Float16*      whq  = (_Float16*)(ws + 2 * SZ_H + 2 * SZ_B + SZ_KV + SZ_SCL);
    _Float16*      whk  = whq + (size_t)C_DIM * C_DIM;
    _Float16*      whv  = whk + (size_t)C_DIM * C_DIM;
    _Float16*      whp  = whv + (size_t)C_DIM * C_DIM;
    float*         out  = (float*)d_out;

    bn_scales_kernel<<<3, 128, 0, stream>>>(qg, qb, qm, qv, scl + 0 * 2 * C_DIM);
    bn_scales_kernel<<<3, 128, 0, stream>>>(kg, kb, km, kv, scl + 1 * 2 * C_DIM);
    bn_scales_kernel<<<3, 128, 0, stream>>>(vg, vb, vm, vv, scl + 2 * 2 * C_DIM);
    bn_scales_kernel<<<3, 128, 0, stream>>>(pg, pb, pm, pv, scl + 3 * 2 * C_DIM);

    wcvt_kernel<<<144, 256, 0, stream>>>(qw, whq);
    wcvt_kernel<<<144, 256, 0, stream>>>(kw, whk);
    wcvt_kernel<<<144, 256, 0, stream>>>(vw, whv);
    wcvt_kernel<<<144, 256, 0, stream>>>(pw, whp);

    quant_tr_kernel<<<dim3(64, 12, 8), 256, 0, stream>>>(x, xs_t);

    qkv_gemm_kernel<<<dim3(16, 24, 8), 256, 0, stream>>>(xs_t, whq, whk, whv, scl, qt, k8, v8p);

    kv_gemm_kernel<<<dim3(9, 64), 32, 0, stream>>>(k8, v8p, kvT);

    attn_gemm_kernel<<<dim3(24, 64), 256, 0, stream>>>(qt, kvT, xs_t);

    pconv_gemm_kernel<<<dim3(16, 24, 8), 256, 0, stream>>>(xs_t, whp, scl + 3 * 2 * C_DIM, out);
}
